// GAT_Net_57191784513885
// MI455X (gfx1250) — compile-verified, matrix operand order checked by reading, not run on hardware
//
#include <hip/hip_runtime.h>
#include <hip/hip_bf16.h>

// GAT (2-layer) for gfx1250: bf16 WMMA GEMMs (fragment-swizzled LDS) + atomic
// segment-softmax edge kernels.

#define F_IN      256
#define HEADS     8
#define HID       16
#define HD1       128   // HEADS*HID
#define NCLS      16
#define NEG_SLOPE 0.2f

typedef __attribute__((ext_vector_type(16))) __bf16 v16bf;
typedef __attribute__((ext_vector_type(8)))  __bf16 v8bf;
typedef __attribute__((ext_vector_type(4)))  __bf16 v4bf;
typedef __attribute__((ext_vector_type(8)))  float  v8f;

#define SHUF16(lo, hi) __builtin_shufflevector(lo, hi, 0, 1, 2, 3, 4, 5, 6, 7, \
                                               8, 9, 10, 11, 12, 13, 14, 15)

// ---- monotonic float <-> uint key (for atomicMax-based segment max) ----
__device__ __forceinline__ unsigned fkey(float f) {
    unsigned u = __float_as_uint(f);
    return (u & 0x80000000u) ? ~u : (u | 0x80000000u);
}
__device__ __forceinline__ float funkey(unsigned k) {
    unsigned u = (k & 0x80000000u) ? (k & 0x7FFFFFFFu) : ~k;
    return __uint_as_float(u);
}

// B-fragment element mapping within a 32-wide K step (16-bit B-matrix layout):
// lanes 0-15 hold K in {0..7, 16..23}; lanes 16-31 hold K in {8..15, 24..31}.
//   kk (0..31) -> hiBit = (kk>>3)&1 ; elem i = (kk&7) + ((kk>>4)<<3)

// =====================  GEMM1: h1[N,128] = feat[N,256] @ W1[256,128]  =====================
// 256 threads = 8 waves; block covers 32 rows; wave w covers cols [16w,16w+16).
__global__ __launch_bounds__(256)
void gat_gemm1(const float* __restrict__ feat, const float* __restrict__ W1,
               float* __restrict__ h1, int nNodes)
{
    // A: natural [row][K] layout (fragment halves are contiguous 16B chunks).
    __shared__ __attribute__((aligned(32))) __bf16 sA[32 * F_IN];   // 16 KB
    // B: fragment-swizzled. slot = kstep*256 + tile*32 + lane; 16 contiguous bf16/slot.
    __shared__ __attribute__((aligned(32))) __bf16 sB[64 * HD1];    // 16 KB

    const int tid = threadIdx.x;
    const int rowBase = blockIdx.x * 32;

    // Stage A: float4 global read -> v4bf packed LDS store (8 per thread).
    for (int i4 = tid; i4 < (32 * F_IN) / 4; i4 += 256) {
        int i = i4 * 4;
        int r = i >> 8, k = i & 255;
        int row = rowBase + r;
        float4 f = (row < nNodes) ? *(const float4*)&feat[(size_t)row * F_IN + k]
                                  : make_float4(0.f, 0.f, 0.f, 0.f);
        v4bf v;
        v[0] = (__bf16)f.x; v[1] = (__bf16)f.y; v[2] = (__bf16)f.z; v[3] = (__bf16)f.w;
        *(v4bf*)&sA[i] = v;
    }

    const int lane = tid & 31;
    const int wave = tid >> 5;
    const int lr = lane & 15;
    const int hi = (lane >> 4) << 3;   // 0 for lanes 0-15, 8 for lanes 16-31

    v8f c0 = {};
    v8f c1 = {};

    for (int kc = 0; kc < F_IN; kc += 64) {
        __syncthreads();               // previous chunk fully consumed
        // Stage B swizzled: 512 slots, 2 per thread; one 32B LDS store each.
        for (int slot = tid; slot < 512; slot += 256) {
            int kstep = slot >> 8;          // 0..1  (which 32-wide K step)
            int tile  = (slot >> 5) & 7;    // 0..7  (16-col tile)
            int ln    = slot & 31;          // consuming lane
            int lrr   = ln & 15;
            int hib   = (ln >> 4) << 3;
            v16bf v;
#pragma unroll
            for (int i = 0; i < 8; ++i) {
                int kk  = kc + kstep * 32 + hib + i;
                int kk2 = kc + kstep * 32 + 16 + hib + i;
                v[i]     = (__bf16)W1[(size_t)kk  * HD1 + tile * 16 + lrr];
                v[i + 8] = (__bf16)W1[(size_t)kk2 * HD1 + tile * 16 + lrr];
            }
            *(v16bf*)&sB[slot * 16] = v;
        }
        __syncthreads();
#pragma unroll
        for (int k0 = 0; k0 < 64; k0 += 32) {
            v16bf b = *(const v16bf*)&sB[((((k0 >> 5) << 3) + wave) * 32 + lane) * 16];
            const int abase = lr * F_IN + kc + k0 + hi;
            v8bf a0lo = *(const v8bf*)&sA[abase];
            v8bf a0hi = *(const v8bf*)&sA[abase + 16];
            v8bf a1lo = *(const v8bf*)&sA[abase + 16 * F_IN];
            v8bf a1hi = *(const v8bf*)&sA[abase + 16 * F_IN + 16];
            v16bf a0 = SHUF16(a0lo, a0hi);
            v16bf a1 = SHUF16(a1lo, a1hi);
            c0 = __builtin_amdgcn_wmma_f32_16x16x32_bf16(false, a0, false, b, (short)0, c0, false, false);
            c1 = __builtin_amdgcn_wmma_f32_16x16x32_bf16(false, a1, false, b, (short)0, c1, false, false);
        }
    }

    const int colBase = wave * 16;
    const int mOff = (lane >> 4) << 3;
#pragma unroll
    for (int v = 0; v < 8; ++v) {
        int m  = v + mOff;
        int r0 = rowBase + m;
        int r1 = rowBase + 16 + m;
        if (r0 < nNodes) h1[(size_t)r0 * HD1 + colBase + lr] = c0[v];
        if (r1 < nNodes) h1[(size_t)r1 * HD1 + colBase + lr] = c1[v];
    }
}

// =====================  GEMM2: h2[N,16] = x2[N,128] @ W2[128,16]  =====================
// 64 threads = 2 waves; block covers 32 rows; each wave one 16-row tile, full 16 cols.
__global__ __launch_bounds__(64)
void gat_gemm2(const float* __restrict__ x2, const float* __restrict__ W2,
               float* __restrict__ h2, int nNodes)
{
    __shared__ __attribute__((aligned(32))) __bf16 sA[32 * HD1];    // 8 KB
    // B swizzled: slot = kstep*32 + lane (4 K-steps); 16 contiguous bf16/slot.
    __shared__ __attribute__((aligned(32))) __bf16 sB[HD1 * NCLS];  // 4 KB

    const int tid = threadIdx.x;
    const int rowBase = blockIdx.x * 32;

    // Stage B swizzled: 128 slots, 2 per thread.
    for (int slot = tid; slot < 128; slot += 64) {
        int kstep = slot >> 5;
        int ln    = slot & 31;
        int lrr   = ln & 15;
        int hib   = (ln >> 4) << 3;
        v16bf v;
#pragma unroll
        for (int i = 0; i < 8; ++i) {
            int kk  = kstep * 32 + hib + i;
            int kk2 = kstep * 32 + 16 + hib + i;
            v[i]     = (__bf16)W2[kk  * NCLS + lrr];
            v[i + 8] = (__bf16)W2[kk2 * NCLS + lrr];
        }
        *(v16bf*)&sB[slot * 16] = v;
    }
    // Stage A: float4 -> v4bf (16 per thread).
    for (int i4 = tid; i4 < (32 * HD1) / 4; i4 += 64) {
        int i = i4 * 4;
        int r = i >> 7, k = i & 127;
        int row = rowBase + r;
        float4 f = (row < nNodes) ? *(const float4*)&x2[(size_t)row * HD1 + k]
                                  : make_float4(0.f, 0.f, 0.f, 0.f);
        v4bf v;
        v[0] = (__bf16)f.x; v[1] = (__bf16)f.y; v[2] = (__bf16)f.z; v[3] = (__bf16)f.w;
        *(v4bf*)&sA[i] = v;
    }
    __syncthreads();

    const int lane = tid & 31;
    const int wave = tid >> 5;
    const int lr = lane & 15;
    const int hi = (lane >> 4) << 3;
    const int aRow = wave * 16 + lr;

    v8f c = {};
#pragma unroll
    for (int k0 = 0; k0 < HD1; k0 += 32) {
        v16bf b = *(const v16bf*)&sB[((k0 >> 5) * 32 + lane) * 16];
        const int abase = aRow * HD1 + k0 + hi;
        v8bf alo = *(const v8bf*)&sA[abase];
        v8bf ahi = *(const v8bf*)&sA[abase + 16];
        v16bf a = SHUF16(alo, ahi);
        c = __builtin_amdgcn_wmma_f32_16x16x32_bf16(false, a, false, b, (short)0, c, false, false);
    }

    const int mOff = (lane >> 4) << 3;
#pragma unroll
    for (int v = 0; v < 8; ++v) {
        int row = rowBase + wave * 16 + v + mOff;
        if (row < nNodes) h2[(size_t)row * NCLS + lr] = c[v];
    }
}

// =====================  Attention scores: el/er[n,H] = <h[n,h,:], al/ar[h,:]>  =====================
template <int H, int D>
__global__ void attn_scores(const float* __restrict__ hmat, const float* __restrict__ al,
                            const float* __restrict__ ar, float* __restrict__ el,
                            float* __restrict__ er, int n)
{
    int i = blockIdx.x * blockDim.x + threadIdx.x;
    if (i >= n * H) return;
    int h = i % H;
    const float* hp = hmat + (size_t)(i / H) * (H * D) + h * D;
    float sl = 0.f, sr = 0.f;
#pragma unroll
    for (int d = 0; d < D; ++d) {
        float v = hp[d];
        sl += v * al[h * D + d];
        sr += v * ar[h * D + d];
    }
    el[i] = sl;
    er[i] = sr;
}

// =====================  Edge pass 1: segment max of leaky_relu(el[src]+er[dst])  =====================
template <int H>
__global__ void edge_lrelu_max(const int* __restrict__ src, const int* __restrict__ dst,
                               const float* __restrict__ el, const float* __restrict__ er,
                               unsigned* __restrict__ nmax, int nE)
{
    int i = blockIdx.x * blockDim.x + threadIdx.x;
    if (i >= nE * H) return;
    int e = i / H, h = i % H;
    int s = src[e], d = dst[e];
    float x = el[s * H + h] + er[d * H + h];
    x = (x > 0.f) ? x : NEG_SLOPE * x;
    atomicMax(&nmax[d * H + h], fkey(x));
}

// =====================  Edge pass 2: denom = segment sum of exp(e - max)  =====================
template <int H>
__global__ void edge_exp_sum(const int* __restrict__ src, const int* __restrict__ dst,
                             const float* __restrict__ el, const float* __restrict__ er,
                             const unsigned* __restrict__ nmax, float* __restrict__ denom, int nE)
{
    int i = blockIdx.x * blockDim.x + threadIdx.x;
    if (i >= nE * H) return;
    int e = i / H, h = i % H;
    int s = src[e], d = dst[e];
    float x = el[s * H + h] + er[d * H + h];
    x = (x > 0.f) ? x : NEG_SLOPE * x;
    atomicAdd(&denom[d * H + h], __expf(x - funkey(nmax[d * H + h])));
}

// =====================  Edge pass 3: acc[dst] += a * h[src]  (one thread per edge-feature)  ===========
template <int H, int D>
__global__ void edge_scatter(const int* __restrict__ src, const int* __restrict__ dst,
                             const float* __restrict__ el, const float* __restrict__ er,
                             const unsigned* __restrict__ nmax, const float* __restrict__ denom,
                             const float* __restrict__ hsrc, float* __restrict__ acc, int nE)
{
    long long i = (long long)blockIdx.x * blockDim.x + threadIdx.x;
    if (i >= (long long)nE * (H * D)) return;
    int e = (int)(i / (H * D));
    int j = (int)(i % (H * D));
    int h = j / D;
    int s = src[e], d = dst[e];
    float x = el[s * H + h] + er[d * H + h];
    x = (x > 0.f) ? x : NEG_SLOPE * x;
    float a = __expf(x - funkey(nmax[d * H + h])) / denom[d * H + h];
    atomicAdd(&acc[(size_t)d * (H * D) + j], a * hsrc[(size_t)s * (H * D) + j]);
}

// =====================  Bias + ELU  =====================
__global__ void bias_elu(const float* __restrict__ acc, const float* __restrict__ b,
                         float* __restrict__ out, int total, int cols)
{
    int i = blockIdx.x * blockDim.x + threadIdx.x;
    if (i >= total) return;
    float v = acc[i] + b[i % cols];
    out[i] = (v > 0.f) ? v : (__expf(v) - 1.f);
}

// =====================  Bias + log_softmax (16 classes)  =====================
__global__ void bias_logsoftmax(const float* __restrict__ acc, const float* __restrict__ b,
                                float* __restrict__ out, int n)
{
    int node = blockIdx.x * blockDim.x + threadIdx.x;
    if (node >= n) return;
    float v[NCLS];
    float m = -3.0e38f;
#pragma unroll
    for (int c = 0; c < NCLS; ++c) {
        v[c] = acc[(size_t)node * NCLS + c] + b[c];
        m = fmaxf(m, v[c]);
    }
    float s = 0.f;
#pragma unroll
    for (int c = 0; c < NCLS; ++c) s += __expf(v[c] - m);
    float lse = __logf(s) + m;
#pragma unroll
    for (int c = 0; c < NCLS; ++c) out[(size_t)node * NCLS + c] = v[c] - lse;
}

// =====================  Launch  =====================
extern "C" void kernel_launch(void* const* d_in, const int* in_sizes, int n_in,
                              void* d_out, int out_size, void* d_ws, size_t ws_size,
                              hipStream_t stream)
{
    const float* feat = (const float*)d_in[0];
    const int*   src  = (const int*)d_in[1];
    const int*   dst  = (const int*)d_in[2];
    const float* W1   = (const float*)d_in[3];
    const float* al1  = (const float*)d_in[4];
    const float* ar1  = (const float*)d_in[5];
    const float* b1   = (const float*)d_in[6];
    const float* W2   = (const float*)d_in[7];
    const float* al2  = (const float*)d_in[8];
    const float* ar2  = (const float*)d_in[9];
    const float* b2   = (const float*)d_in[10];
    float* out = (float*)d_out;

    const int N = in_sizes[0] / F_IN;
    const int E = in_sizes[1];

    // ---- workspace carve-up ----
    char* ws = (char*)d_ws;
    size_t off = 0;
    auto take = [&](size_t bytes) -> void* {
        void* p = ws + off;
        off = (off + bytes + 255) & ~(size_t)255;
        return p;
    };
    float*    acc1   = (float*)   take((size_t)N * HD1 * 4);
    float*    h1     = (float*)   take((size_t)N * HD1 * 4);  // reused as x2 after ELU
    float*    el1    = (float*)   take((size_t)N * HEADS * 4);
    float*    er1    = (float*)   take((size_t)N * HEADS * 4);
    unsigned* nmax1  = (unsigned*)take((size_t)N * HEADS * 4);
    float*    denom1 = (float*)   take((size_t)N * HEADS * 4);
    float*    h2     = (float*)   take((size_t)N * NCLS * 4);
    float*    el2    = (float*)   take((size_t)N * 4);
    float*    er2    = (float*)   take((size_t)N * 4);
    unsigned* nmax2  = (unsigned*)take((size_t)N * 4);
    float*    denom2 = (float*)   take((size_t)N * 4);
    float*    acc2   = (float*)   take((size_t)N * NCLS * 4);
    (void)ws_size; (void)n_in; (void)out_size;

    // ---- zero accumulators / max keys (key 0 == -inf sentinel) ----
    hipMemsetAsync(acc1,   0, (size_t)N * HD1 * 4,   stream);
    hipMemsetAsync(nmax1,  0, (size_t)N * HEADS * 4, stream);
    hipMemsetAsync(denom1, 0, (size_t)N * HEADS * 4, stream);
    hipMemsetAsync(nmax2,  0, (size_t)N * 4,         stream);
    hipMemsetAsync(denom2, 0, (size_t)N * 4,         stream);
    hipMemsetAsync(acc2,   0, (size_t)N * NCLS * 4,  stream);

    const int rowBlocks = (N + 31) / 32;

    // ---- Layer 1 ----
    gat_gemm1<<<rowBlocks, 256, 0, stream>>>(feat, W1, h1, N);
    attn_scores<HEADS, HID><<<(N * HEADS + 255) / 256, 256, 0, stream>>>(h1, al1, ar1, el1, er1, N);
    edge_lrelu_max<HEADS><<<(E * HEADS + 255) / 256, 256, 0, stream>>>(src, dst, el1, er1, nmax1, E);
    edge_exp_sum<HEADS><<<(E * HEADS + 255) / 256, 256, 0, stream>>>(src, dst, el1, er1, nmax1, denom1, E);
    {
        long long work = (long long)E * HD1;
        int blocks = (int)((work + 255) / 256);
        edge_scatter<HEADS, HID><<<blocks, 256, 0, stream>>>(src, dst, el1, er1, nmax1, denom1, h1, acc1, E);
    }
    bias_elu<<<(N * HD1 + 255) / 256, 256, 0, stream>>>(acc1, b1, h1 /*x2*/, N * HD1, HD1);

    // ---- Layer 2 ----
    gat_gemm2<<<rowBlocks, 64, 0, stream>>>(h1 /*x2*/, W2, h2, N);
    attn_scores<1, NCLS><<<(N + 255) / 256, 256, 0, stream>>>(h2, al2, ar2, el2, er2, N);
    edge_lrelu_max<1><<<(E + 255) / 256, 256, 0, stream>>>(src, dst, el2, er2, nmax2, E);
    edge_exp_sum<1><<<(E + 255) / 256, 256, 0, stream>>>(src, dst, el2, er2, nmax2, denom2, E);
    {
        long long work = (long long)E * NCLS;
        int blocks = (int)((work + 255) / 256);
        edge_scatter<1, NCLS><<<blocks, 256, 0, stream>>>(src, dst, el2, er2, nmax2, denom2, h2, acc2, E);
    }
    bias_logsoftmax<<<(N + 255) / 256, 256, 0, stream>>>(acc2, b2, out, N);
}
